// GNN1_61924838474117
// MI455X (gfx1250) — matrix operand order repaired and checked by reading, not compiled, not run on hardware
//
#include <hip/hip_runtime.h>
#include <stdint.h>

// ---------------- problem constants ----------------
#define BATCH 256
#define NG0   32
#define FS    30
#define FSP   32          // K padded to 32 for WMMA
#define DD    1024
#define HHN   3
#define HC    3072
#define EPG   160
#define ETOT  (BATCH*EPG) // 40960
#define NMAX  (BATCH*NG0) // 8192
#define KP1   26
#define KP2   13
#define KP3   4

// CDNA5 async global->LDS copies (ASYNCcnt). Set to 0 if the assembler rejects
// the mnemonics; the fallback path keeps the identical pipeline structure.
#ifndef USE_ASYNC_LDS
#define USE_ASYNC_LDS 1
#endif

// ---------------- types ----------------
typedef __bf16 bf16_t;
typedef __attribute__((ext_vector_type(16))) bf16_t v16bf;
typedef __attribute__((ext_vector_type(8)))  bf16_t v8bf;
typedef __attribute__((ext_vector_type(8)))  float  v8f;
typedef __attribute__((ext_vector_type(8)))  unsigned short v8us;

union FragBF { v16bf v; v8bf h[2]; };

__device__ __forceinline__ unsigned short f2bf(float f) {
  unsigned u = __float_as_uint(f);
  u += 0x7FFFu + ((u >> 16) & 1u);          // round-to-nearest-even
  return (unsigned short)(u >> 16);
}
__device__ __forceinline__ float bf2f(unsigned short h) {
  return __uint_as_float(((unsigned)h) << 16);
}
// order-preserving float<->uint encoding for atomic max
__device__ __forceinline__ unsigned fenc(float f) {
  unsigned u = __float_as_uint(f);
  return (u & 0x80000000u) ? ~u : (u | 0x80000000u);
}
__device__ __forceinline__ float fdec(unsigned u) {
  return (u & 0x80000000u) ? __uint_as_float(u & 0x7FFFFFFFu) : __uint_as_float(~u);
}

// 16-byte global->LDS copy: async DMA on CDNA5, sync fallback otherwise.
__device__ __forceinline__ void cp16(const unsigned short* g, unsigned short* l) {
#if USE_ASYNC_LDS
  unsigned loff = (unsigned)(size_t)l;                 // low 32 bits = LDS offset
  unsigned long long ga = (unsigned long long)(size_t)g;
  asm volatile("global_load_async_to_lds_b128 %0, %1, off"
               :: "v"(loff), "v"(ga) : "memory");
#else
  *(v8us*)l = *(const v8us*)g;
#endif
}
__device__ __forceinline__ void wait_async_lds() {
#if USE_ASYNC_LDS
  asm volatile("s_wait_asynccnt 0x0" ::: "memory");
#endif
}

// ---------------- WMMA GEMM: C[MxN] = A[MxK](bf16) * Bt[NxK](bf16) (+bias)(+relu) -------------
// Bt is PRE-TRANSPOSED (N-major): both tiles stream to LDS as straight b128 copies.
// block tile 128x128, K-step 32, 8 waves as 2(m) x 4(n), wave tile 64x32 = 4x2 WMMA accums.
// Double-buffered LDS + async copies: DMA of tile t+1 overlaps WMMA of tile t.
#define BM 128
#define BN 128
#define BKK 32
#define LSTR 40   // LDS row stride (bf16 elems): 80B -> conflict-free b128 frag loads

__global__ __launch_bounds__(256) void k_gemm(
    const unsigned short* __restrict__ A, const unsigned short* __restrict__ Bt,
    const float* __restrict__ bias, float* __restrict__ outF,
    unsigned short* __restrict__ outH, int M, int N, int K, int relu)
{
  __shared__ unsigned short lA[2][BM * LSTR];
  __shared__ unsigned short lB[2][BN * LSTR];   // [n][k]
  const int tid  = threadIdx.x;
  const int lane = tid & 31;
  const int wid  = tid >> 5;
  const int wm   = wid >> 2;                    // 0..1
  const int wn   = wid & 3;                     // 0..3
  const int m0   = blockIdx.y * BM;
  const int n0   = blockIdx.x * BN;

  // two 16B chunks per thread per tile (A and B each): ch -> row r, k-offset c8
  const int r0 = tid >> 2,          c80 = (tid & 3) << 3;         // ch = tid
  const int r1 = (tid + 256) >> 2,  c81 = ((tid + 256) & 3) << 3; // ch = tid+256

  v8f acc[4][2];
#pragma unroll
  for (int i = 0; i < 4; i++)
#pragma unroll
    for (int j = 0; j < 2; j++)
#pragma unroll
      for (int r = 0; r < 8; r++) acc[i][j][r] = 0.f;

  const int ntiles = K / BKK;

  // prologue: tile 0 into buffer 0
  {
    const int k0 = 0;
    cp16(&A [(size_t)(m0 + r0) * K + k0 + c80], &lA[0][r0 * LSTR + c80]);
    cp16(&A [(size_t)(m0 + r1) * K + k0 + c81], &lA[0][r1 * LSTR + c81]);
    cp16(&Bt[(size_t)(n0 + r0) * K + k0 + c80], &lB[0][r0 * LSTR + c80]);
    cp16(&Bt[(size_t)(n0 + r1) * K + k0 + c81], &lB[0][r1 * LSTR + c81]);
  }

  int p = 0;
  for (int t = 0; t < ntiles; ++t) {
    wait_async_lds();          // my async copies for tile t complete
    __syncthreads();           // everyone's copies visible; prior reads of buf p^1 done
    if (t + 1 < ntiles) {      // kick DMA for tile t+1 into the other buffer
      const int k0 = (t + 1) * BKK;
      cp16(&A [(size_t)(m0 + r0) * K + k0 + c80], &lA[p ^ 1][r0 * LSTR + c80]);
      cp16(&A [(size_t)(m0 + r1) * K + k0 + c81], &lA[p ^ 1][r1 * LSTR + c81]);
      cp16(&Bt[(size_t)(n0 + r0) * K + k0 + c80], &lB[p ^ 1][r0 * LSTR + c80]);
      cp16(&Bt[(size_t)(n0 + r1) * K + k0 + c81], &lB[p ^ 1][r1 * LSTR + c81]);
    }

    // fragments per ISA 16-bit A 16x32 layout:
    //   lanes 0-15: row=lane, K 0..7 & 16..23 ; lanes 16-31: row=lane-16, K 8..15 & 24..31
    FragBF a[4], b[2];
    const int arow = lane & 15;
    const int agrp = (lane >> 4) << 3;   // 0 or 8
#pragma unroll
    for (int i = 0; i < 4; i++) {
      int r = wm * 64 + i * 16 + arow;
      a[i].h[0] = *(const v8bf*)&lA[p][r * LSTR + agrp];
      a[i].h[1] = *(const v8bf*)&lA[p][r * LSTR + 16 + agrp];
    }
    const int kb = (lane >> 4) << 3;     // 0 or 8
#pragma unroll
    for (int j = 0; j < 2; j++) {
      int cN = wn * 32 + j * 16 + (lane & 15);
      b[j].h[0] = *(const v8bf*)&lB[p][cN * LSTR + kb];
      b[j].h[1] = *(const v8bf*)&lB[p][cN * LSTR + kb + 16];
    }
#pragma unroll
    for (int i = 0; i < 4; i++)
#pragma unroll
      for (int j = 0; j < 2; j++)
        acc[i][j] = __builtin_amdgcn_wmma_f32_16x16x32_bf16(
            false, a[i].v, false, b[j].v, (short)0, acc[i][j], false, false);
    __syncthreads();           // all waves done reading buf p
    p ^= 1;
  }

  // epilogue: C/D layout -> VGPR r: M = base+r (lanes 0-15) / base+8+r (lanes 16-31), N = lane&15
  const int roff = (lane >> 4) << 3;
  const int coln = lane & 15;
#pragma unroll
  for (int j = 0; j < 2; j++) {
    int gn = n0 + wn * 32 + j * 16 + coln;
    float bv = bias ? bias[gn] : 0.f;
#pragma unroll
    for (int i = 0; i < 4; i++) {
      int gmb = m0 + wm * 64 + i * 16 + roff;
#pragma unroll
      for (int r = 0; r < 8; r++) {
        float vv = acc[i][j][r] + bv;
        if (relu) vv = fmaxf(vv, 0.f);
        size_t o = (size_t)(gmb + r) * N + gn;
        if (outF) outF[o] = vv;
        if (outH) outH[o] = f2bf(vv);
      }
    }
  }
}

// ---------------- elementwise helpers ----------------
__global__ void k_fill_f32(float* p, float v, int n) {
  int t = blockIdx.x * 256 + threadIdx.x; if (t < n) p[t] = v;
}
__global__ void k_fill_i32(int* p, int v, int n) {
  int t = blockIdx.x * 256 + threadIdx.x; if (t < n) p[t] = v;
}
__global__ void k_convert(const float* __restrict__ s, unsigned short* __restrict__ d, int n) {
  int t = blockIdx.x * 256 + threadIdx.x; if (t < n) d[t] = f2bf(s[t]);
}
// convert + transpose: W[KxN] fp32 -> Wt[NxK] bf16, 32x32 LDS tile, coalesced both sides
__global__ __launch_bounds__(256) void k_convt(const float* __restrict__ s,
                                               unsigned short* __restrict__ d,
                                               int K, int N) {
  __shared__ unsigned short t[32][33];
  int n0 = blockIdx.x * 32, k0 = blockIdx.y * 32;
  int tx = threadIdx.x & 31, ty = threadIdx.x >> 5;   // 32x8
  for (int r = ty; r < 32; r += 8)
    t[r][tx] = f2bf(s[(size_t)(k0 + r) * N + n0 + tx]);
  __syncthreads();
  for (int r = ty; r < 32; r += 8)
    d[(size_t)(n0 + r) * K + k0 + tx] = t[tx][r];
}
// layer-1 W: [30 x 3072] fp32 -> transposed padded [3072 x 32] bf16
__global__ void k_padt(const float* __restrict__ s, unsigned short* __restrict__ d) {
  int t = blockIdx.x * 256 + threadIdx.x;
  if (t >= HC * FSP) return;
  int n = t / FSP, k = t - n * FSP;
  d[t] = f2bf(k < FS ? s[(size_t)k * HC + n] : 0.f);
}
// x: [8192 x 30] fp32 -> padded [8192 x 32] bf16
__global__ void k_pad_cols(const float* __restrict__ s, unsigned short* __restrict__ d,
                           int rows, int scols, int dcols) {
  int t = blockIdx.x * 256 + threadIdx.x;
  if (t >= rows * dcols) return;
  int r = t / dcols, c = t - r * dcols;
  d[t] = f2bf(c < scols ? s[(size_t)r * scols + c] : 0.f);
}
__global__ void k_edges_init(const int* __restrict__ ei, int* s, int* d, int* m) {
  int t = blockIdx.x * 256 + threadIdx.x;
  if (t >= ETOT) return;
  s[t] = ei[t]; d[t] = ei[ETOT + t]; m[t] = 1;
}

// ---------------- attention coefficients: al_s/al_d = <xl[n,h,:], a[h,:]> ----------------
__global__ __launch_bounds__(256) void k_al(
    const unsigned short* __restrict__ xl, const float* __restrict__ asrc,
    const float* __restrict__ adst, float* __restrict__ alS, float* __restrict__ alD)
{
  __shared__ float red[256];
  int n = blockIdx.x, tid = threadIdx.x;
  for (int h = 0; h < HHN; h++) {
    float ps = 0.f, pd = 0.f;
    for (int c = tid; c < DD; c += 256) {
      float v = bf2f(xl[(size_t)n * HC + h * DD + c]);
      ps += v * asrc[h * DD + c];
      pd += v * adst[h * DD + c];
    }
    red[tid] = ps; __syncthreads();
    for (int s = 128; s > 0; s >>= 1) { if (tid < s) red[tid] += red[tid + s]; __syncthreads(); }
    if (tid == 0) alS[n * HHN + h] = red[0];
    __syncthreads();
    red[tid] = pd; __syncthreads();
    for (int s = 128; s > 0; s >>= 1) { if (tid < s) red[tid] += red[tid + s]; __syncthreads(); }
    if (tid == 0) alD[n * HHN + h] = red[0];
    __syncthreads();
  }
}

// ---------------- per-graph segment softmax over in-edges (incl. self loops) ----------------
__global__ __launch_bounds__(256) void k_attn(
    const int* __restrict__ src, const int* __restrict__ dst, const int* __restrict__ msk,
    const float* __restrict__ alS, const float* __restrict__ alD,
    float* __restrict__ alpha, int n_per)
{
  __shared__ unsigned mxe[NG0 * HHN];
  __shared__ float    den[NG0 * HHN];
  int b = blockIdx.x, tid = threadIdx.x;
  int nb = b * n_per, eb = b * EPG;
  int tot = (EPG + n_per) * HHN;
  if (tid < NG0 * HHN) { mxe[tid] = 0u; den[tid] = 0.f; }
  __syncthreads();

  float lg[3]; int dh[3]; int aix[3]; int mm[3]; int cnt = 0;
  for (int t = tid; t < tot; t += 256) {
    int e = t / HHN, h = t - e * HHN;
    int s, d_, m, ai;
    if (e < EPG) { int ge = eb + e; s = src[ge]; d_ = dst[ge]; m = msk[ge]; ai = ge * HHN + h; }
    else { int i = e - EPG; s = nb + i; d_ = nb + i; m = 1; ai = (ETOT + nb + i) * HHN + h; }
    float l = alS[s * HHN + h] + alD[d_ * HHN + h];
    l = l > 0.f ? l : 0.2f * l;                       // leaky_relu(0.2)
    int dl = m ? ((d_ - nb) * HHN + h) : 0;
    if (m) atomicMax(&mxe[dl], fenc(l));
    lg[cnt] = l; dh[cnt] = dl; aix[cnt] = ai; mm[cnt] = m; cnt++;
  }
  __syncthreads();
  float ex[3];
  for (int q = 0; q < cnt; q++) {
    float e_ = mm[q] ? __expf(lg[q] - fdec(mxe[dh[q]])) : 0.f;
    ex[q] = e_;
    if (mm[q]) atomicAdd(&den[dh[q]], e_);
  }
  __syncthreads();
  for (int q = 0; q < cnt; q++) {
    float dn = den[dh[q]];
    alpha[aix[q]] = (mm[q] && dn > 0.f) ? ex[q] / dn : 0.f;
  }
}

// ---------------- aggregation: out[n] = sum_e alpha[e]*xl[src[e]] + bias (per 256-ch chunk) ----
__global__ __launch_bounds__(256) void k_agg(
    const unsigned short* __restrict__ xl,
    const int* __restrict__ src, const int* __restrict__ dst, const int* __restrict__ msk,
    const float* __restrict__ alpha, const float* __restrict__ bias,
    unsigned short* __restrict__ outBf, int n_per)
{
  __shared__ float acc[NG0 * 256];      // column tid is private to each thread: no atomics
  int chunk = blockIdx.x, b = blockIdx.y, tid = threadIdx.x;
  int cbase = chunk * 256;
  int h = chunk >> 2;                   // 4 chunks of 256 per head
  int nb = b * n_per, eb = b * EPG;
#pragma unroll
  for (int i = 0; i < NG0; i++) acc[i * 256 + tid] = 0.f;

  for (int e = 0; e < EPG; e++) {
    int ge = eb + e;
    int m = msk[ge];
    if (m) {
      int s = src[ge], d_ = dst[ge];
      float a = alpha[ge * HHN + h];
      float v = bf2f(xl[(size_t)s * HC + cbase + tid]);
      acc[(d_ - nb) * 256 + tid] += a * v;
    }
  }
  for (int i = 0; i < n_per; i++) {     // self loops
    int node = nb + i;
    float a = alpha[(ETOT + node) * HHN + h];
    float v = bf2f(xl[(size_t)node * HC + cbase + tid]);
    acc[i * 256 + tid] += a * v;
  }
  float bv = bias[cbase + tid];
  for (int i = 0; i < n_per; i++)
    outBf[(size_t)(nb + i) * HC + cbase + tid] = f2bf(acc[i * 256 + tid] + bv);
}

// ---------------- top-k pooling pieces ----------------
__global__ __launch_bounds__(256) void k_score(
    const float* __restrict__ hF, const float* __restrict__ p, float* __restrict__ score)
{
  __shared__ float red[256];
  int n = blockIdx.x, tid = threadIdx.x;
  float s1 = 0.f, s2 = 0.f;
  for (int c = tid; c < DD; c += 256) { float pv = p[c]; s1 += hF[(size_t)n * DD + c] * pv; s2 += pv * pv; }
  red[tid] = s1; __syncthreads();
  for (int s = 128; s > 0; s >>= 1) { if (tid < s) red[tid] += red[tid + s]; __syncthreads(); }
  s1 = red[0]; __syncthreads();
  red[tid] = s2; __syncthreads();
  for (int s = 128; s > 0; s >>= 1) { if (tid < s) red[tid] += red[tid + s]; __syncthreads(); }
  if (tid == 0) score[n] = tanhf(s1 / sqrtf(red[0]));
}

// one wave32 per graph; selection via cross-lane argmax (k iterations)
__global__ void k_topk(const float* __restrict__ score, int n_per, int kkeep,
                       int* __restrict__ perm, int* __restrict__ new_pos)
{
  int b = blockIdx.x, lane = threadIdx.x;
  int nb = b * n_per;
  float v = (lane < n_per) ? score[nb + lane] : -3.4e38f;
  for (int j = 0; j < kkeep; j++) {
    float bv = v; int bi = lane;
    for (int off = 16; off > 0; off >>= 1) {
      float ov = __shfl_xor(bv, off);
      int   oi = __shfl_xor(bi, off);
      if (ov > bv || (ov == bv && oi < bi)) { bv = ov; bi = oi; }
    }
    if (lane == 0) { perm[b * kkeep + j] = nb + bi; new_pos[nb + bi] = b * kkeep + j; }
    if (lane == bi) v = -3.4e38f;
  }
}

__global__ __launch_bounds__(256) void k_gather(
    const float* __restrict__ hF, const int* __restrict__ perm,
    const float* __restrict__ score, float* __restrict__ hnew,
    unsigned short* __restrict__ actBf)
{
  int row = blockIdx.x, tid = threadIdx.x;
  int old = perm[row];
  float sc = score[old];
#pragma unroll
  for (int g = 0; g < 4; g++) {
    int c = g * 256 + tid;
    float v = hF[(size_t)old * DD + c] * sc;
    hnew[(size_t)row * DD + c] = v;
    actBf[(size_t)row * DD + c] = f2bf(v);
  }
}

__global__ void k_remap(const int* sI, const int* dI, const int* mI, const int* np,
                        int* sO, int* dO, int* mO)
{
  int t = blockIdx.x * 256 + threadIdx.x;
  if (t >= ETOT) return;
  int ns = np[sI[t]], nd = np[dI[t]];
  int m = mI[t] && ns >= 0 && nd >= 0;
  sO[t] = m ? ns : 0; dO[t] = m ? nd : 0; mO[t] = m;
}

__global__ __launch_bounds__(256) void k_mean_add(
    const float* __restrict__ hnew, int kkeep, float* __restrict__ xsum)
{
  int b = blockIdx.x, tid = threadIdx.x;
#pragma unroll
  for (int g = 0; g < 4; g++) {
    int c = g * 256 + tid;
    float s = 0.f;
    for (int r = 0; r < kkeep; r++) s += hnew[(size_t)(b * kkeep + r) * DD + c];
    xsum[(size_t)b * DD + c] += s / (float)kkeep;
  }
}

__global__ __launch_bounds__(256) void k_final(
    const float* __restrict__ z, const float* __restrict__ w2,
    const float* __restrict__ b2, float* __restrict__ out)
{
  __shared__ float red[256];
  int b = blockIdx.x, tid = threadIdx.x;
  float s = 0.f;
  for (int c = tid; c < DD; c += 256) s += z[(size_t)b * DD + c] * w2[c];
  red[tid] = s; __syncthreads();
  for (int st = 128; st > 0; st >>= 1) { if (tid < st) red[tid] += red[tid + st]; __syncthreads(); }
  if (tid == 0) out[b] = red[0] + b2[0];
}

// ---------------- orchestration ----------------
extern "C" void kernel_launch(void* const* d_in, const int* in_sizes, int n_in,
                              void* d_out, int out_size, void* d_ws, size_t ws_size,
                              hipStream_t stream)
{
  (void)in_sizes; (void)n_in; (void)out_size; (void)ws_size;
  const float* x    = (const float*)d_in[0];
  const int*   eidx = (const int*)d_in[2];
  const float *g1W = (const float*)d_in[4],  *g1as = (const float*)d_in[5],
              *g1ad = (const float*)d_in[6],  *g1b = (const float*)d_in[7],
              *t1W = (const float*)d_in[8],  *t1b = (const float*)d_in[9],  *p1 = (const float*)d_in[10];
  const float *g2W = (const float*)d_in[11], *g2as = (const float*)d_in[12],
              *g2ad = (const float*)d_in[13], *g2b = (const float*)d_in[14],
              *t2W = (const float*)d_in[15], *t2b = (const float*)d_in[16], *p2 = (const float*)d_in[17];
  const float *g3W = (const float*)d_in[18], *g3as = (const float*)d_in[19],
              *g3ad = (const float*)d_in[20], *g3b = (const float*)d_in[21],
              *t3W = (const float*)d_in[22], *t3b = (const float*)d_in[23], *p3 = (const float*)d_in[24];
  const float *l1W = (const float*)d_in[25], *l1b = (const float*)d_in[26];
  const float *l2W = (const float*)d_in[27], *l2b = (const float*)d_in[28];
  float* out = (float*)d_out;

  // ---- workspace carve (~190 MB) ----
  uint8_t* ws = (uint8_t*)d_ws; size_t cur = 0;
  auto take = [&](size_t bytes) -> void* {
    void* pp = ws + cur; cur += (bytes + 255) & ~(size_t)255; return pp;
  };
  unsigned short* actsA = (unsigned short*)take((size_t)BATCH * KP1 * DD * 2); // max activation bf16
  unsigned short* wB    = (unsigned short*)take((size_t)HC * DD * 2);          // weights bf16 (transposed)
  unsigned short* xlB   = (unsigned short*)take((size_t)NMAX * HC * 2);
  unsigned short* attB  = (unsigned short*)take((size_t)NMAX * HC * 2);
  float* hF    = (float*)take((size_t)NMAX * DD * 4);
  float* hnew  = (float*)take((size_t)BATCH * KP1 * DD * 4);
  float* alS   = (float*)take((size_t)NMAX * HHN * 4);
  float* alD   = (float*)take((size_t)NMAX * HHN * 4);
  float* alpha = (float*)take((size_t)(ETOT + NMAX) * HHN * 4);
  float* score = (float*)take((size_t)NMAX * 4);
  int* srcA = (int*)take((size_t)ETOT * 4);
  int* dstA = (int*)take((size_t)ETOT * 4);
  int* mskA = (int*)take((size_t)ETOT * 4);
  int* srcB = (int*)take((size_t)ETOT * 4);
  int* dstB = (int*)take((size_t)ETOT * 4);
  int* mskB = (int*)take((size_t)ETOT * 4);
  int* npos = (int*)take((size_t)NMAX * 4);
  int* perm = (int*)take((size_t)BATCH * KP1 * 4);
  float* xsum = (float*)take((size_t)BATCH * DD * 4);

  auto gemm = [&](const unsigned short* Ap, const unsigned short* Btp, const float* bp,
                  float* oF, unsigned short* oH, int M, int N, int K, int relu) {
    dim3 grd(N / 128, M / 128);
    k_gemm<<<grd, 256, 0, stream>>>(Ap, Btp, bp, oF, oH, M, N, K, relu);
  };
  // convert+transpose W[KxN] -> Wt[NxK]
  auto convt = [&](const float* Wp, unsigned short* Wtp, int K, int N) {
    k_convt<<<dim3(N / 32, K / 32), 256, 0, stream>>>(Wp, Wtp, K, N);
  };
#define EW(n) dim3(((n) + 255) / 256)

  k_fill_f32<<<EW(BATCH * DD), 256, 0, stream>>>(xsum, 0.f, BATCH * DD);
  k_edges_init<<<EW(ETOT), 256, 0, stream>>>(eidx, srcA, dstA, mskA);
  k_pad_cols<<<EW(NMAX * FSP), 256, 0, stream>>>(x, actsA, NMAX, FS, FSP);
  k_padt<<<EW(HC * FSP), 256, 0, stream>>>(g1W, wB);     // Wt1: [3072 x 32]

  // ---- layer 1 (N=8192, n_per=32 -> keep 26) ----
  gemm(actsA, wB, nullptr, nullptr, xlB, NMAX, HC, FSP, 0);
  k_al<<<NMAX, 256, 0, stream>>>(xlB, g1as, g1ad, alS, alD);
  k_attn<<<BATCH, 256, 0, stream>>>(srcA, dstA, mskA, alS, alD, alpha, NG0);
  k_agg<<<dim3(HC / 256, BATCH), 256, 0, stream>>>(xlB, srcA, dstA, mskA, alpha, g1b, attB, NG0);
  convt(t1W, wB, HC, DD);                                // Wt: [1024 x 3072]
  gemm(attB, wB, t1b, hF, nullptr, NMAX, DD, HC, 0);
  k_score<<<NMAX, 256, 0, stream>>>(hF, p1, score);
  k_fill_i32<<<EW(NMAX), 256, 0, stream>>>(npos, -1, NMAX);
  k_topk<<<BATCH, 32, 0, stream>>>(score, NG0, KP1, perm, npos);
  k_gather<<<BATCH * KP1, 256, 0, stream>>>(hF, perm, score, hnew, actsA);
  k_remap<<<EW(ETOT), 256, 0, stream>>>(srcA, dstA, mskA, npos, srcB, dstB, mskB);
  k_mean_add<<<BATCH, 256, 0, stream>>>(hnew, KP1, xsum);

  // ---- layer 2 (N=6656, n_per=26 -> keep 13) ----
  const int N2 = BATCH * KP1;
  convt(g2W, wB, DD, HC);                                // Wt: [3072 x 1024]
  gemm(actsA, wB, nullptr, nullptr, xlB, N2, HC, DD, 0);
  k_al<<<N2, 256, 0, stream>>>(xlB, g2as, g2ad, alS, alD);
  k_attn<<<BATCH, 256, 0, stream>>>(srcB, dstB, mskB, alS, alD, alpha, KP1);
  k_agg<<<dim3(HC / 256, BATCH), 256, 0, stream>>>(xlB, srcB, dstB, mskB, alpha, g2b, attB, KP1);
  convt(t2W, wB, HC, DD);
  gemm(attB, wB, t2b, hF, nullptr, N2, DD, HC, 0);
  k_score<<<N2, 256, 0, stream>>>(hF, p2, score);
  k_fill_i32<<<EW(N2), 256, 0, stream>>>(npos, -1, N2);
  k_topk<<<BATCH, 32, 0, stream>>>(score, KP1, KP2, perm, npos);
  k_gather<<<BATCH * KP2, 256, 0, stream>>>(hF, perm, score, hnew, actsA);
  k_remap<<<EW(ETOT), 256, 0, stream>>>(srcB, dstB, mskB, npos, srcA, dstA, mskA);
  k_mean_add<<<BATCH, 256, 0, stream>>>(hnew, KP2, xsum);

  // ---- layer 3 (N=3328, n_per=13 -> keep 4) ----
  const int N3 = BATCH * KP2;
  convt(g3W, wB, DD, HC);
  gemm(actsA, wB, nullptr, nullptr, xlB, N3, HC, DD, 0);
  k_al<<<N3, 256, 0, stream>>>(xlB, g3as, g3ad, alS, alD);
  k_attn<<<BATCH, 256, 0, stream>>>(srcA, dstA, mskA, alS, alD, alpha, KP2);
  k_agg<<<dim3(HC / 256, BATCH), 256, 0, stream>>>(xlB, srcA, dstA, mskA, alpha, g3b, attB, KP2);
  convt(t3W, wB, HC, DD);
  gemm(attB, wB, t3b, hF, nullptr, N3, DD, HC, 0);
  k_score<<<N3, 256, 0, stream>>>(hF, p3, score);
  k_fill_i32<<<EW(N3), 256, 0, stream>>>(npos, -1, N3);
  k_topk<<<BATCH, 32, 0, stream>>>(score, KP2, KP3, perm, npos);
  k_gather<<<BATCH * KP3, 256, 0, stream>>>(hF, perm, score, hnew, actsA);
  k_mean_add<<<BATCH, 256, 0, stream>>>(hnew, KP3, xsum);

  // ---- MLP head ----
  k_convert<<<EW(BATCH * DD), 256, 0, stream>>>(xsum, actsA, BATCH * DD);
  convt(l1W, wB, DD, DD);
  gemm(actsA, wB, l1b, hF, nullptr, BATCH, DD, DD, 1);   // relu fused
  k_final<<<BATCH, 256, 0, stream>>>(hF, l2W, l2b, out);
#undef EW
}
